// MetaNon_local_86947317940386
// MI455X (gfx1250) — matrix operand order repaired and checked by reading, not compiled, not run on hardware
//
#include <hip/hip_runtime.h>

typedef _Float16 v8h  __attribute__((ext_vector_type(8)));
typedef _Float16 v16h __attribute__((ext_vector_type(16)));
typedef float    v8f  __attribute__((ext_vector_type(8)));
typedef float    v4f  __attribute__((ext_vector_type(4)));

union FragU { v16h v; v8h h[2]; };

__device__ inline v8f zero8() { v8f z; for (int i = 0; i < 8; ++i) z[i] = 0.0f; return z; }

// A-matrix fragment (16x32 f16, M rows x K): row-major [row][k], k contiguous.
// lanes 0-15: M=lane, K = k0+0..7 (v0-3) and k0+16..23 (v4-7)
// lanes 16-31: M=lane-16, K = k0+8..15 and k0+24..31
__device__ inline v16h load_a(const _Float16* base, int row0, int k0, int ld) {
  const int lane = threadIdx.x & 31;
  const _Float16* p = base + (size_t)(row0 + (lane & 15)) * ld + k0 + ((lane >> 4) << 3);
  FragU f;
  f.h[0] = *(const v8h*)(p);
  f.h[1] = *(const v8h*)(p + 16);
  return f.v;
}

// B-matrix fragment (32x16 f16, K x N rows): stored row-major [n][k], k contiguous.
// lanes 0-15: N=lane, K = k0+0..7 and k0+16..23 ; lanes 16-31: N=lane-16, K = k0+8..15 and k0+24..31
__device__ inline v16h load_b(const _Float16* base, int row0, int k0, int ld) {
  const int lane = threadIdx.x & 31;
  const _Float16* p = base + (size_t)(row0 + (lane & 15)) * ld + k0 + ((lane >> 4) << 3);
  FragU f;
  f.h[0] = *(const v8h*)(p);
  f.h[1] = *(const v8h*)(p + 16);
  return f.v;
}

__device__ inline v8f wmma16(v16h a, v16h b, v8f c) {
  return __builtin_amdgcn_wmma_f32_16x16x32_f16(false, a, false, b, (short)0, c, false, false);
}

// ---------------- K0: weight cast / concat / BN folding ----------------
__global__ __launch_bounds__(256) void k0_prep(
    const float* __restrict__ w_theta, const float* __restrict__ w_phi,
    const float* __restrict__ w_g, const float* __restrict__ b_theta,
    const float* __restrict__ b_phi, const float* __restrict__ b_g,
    const float* __restrict__ w_out, const float* __restrict__ b_out,
    const float* __restrict__ gamma, const float* __restrict__ beta,
    const float* __restrict__ rmean, const float* __restrict__ rvar,
    _Float16* __restrict__ Wcat, _Float16* __restrict__ Wout16,
    float* __restrict__ bias_cat, float* __restrict__ escale,
    float* __restrict__ eshift) {
  int idx = blockIdx.x * 256 + threadIdx.x;
  if (idx < 384 * 256) {
    int j = idx >> 8, k = idx & 255;
    float v = (j < 128) ? w_theta[j * 256 + k]
            : (j < 256) ? w_phi[(j - 128) * 256 + k]
                        : w_g[(j - 256) * 256 + k];
    Wcat[idx] = (_Float16)v;
  } else if (idx < 384 * 256 + 256 * 128) {
    int i = idx - 384 * 256;
    Wout16[i] = (_Float16)w_out[i];
  } else if (idx < 384 * 256 + 256 * 128 + 384) {
    int j = idx - (384 * 256 + 256 * 128);
    bias_cat[j] = (j < 128) ? b_theta[j] : (j < 256) ? b_phi[j - 128] : b_g[j - 256];
  } else if (idx < 384 * 256 + 256 * 128 + 384 + 256) {
    int o = idx - (384 * 256 + 256 * 128 + 384);
    float inv = gamma[o] * rsqrtf(rvar[o] + 1e-5f);
    escale[o] = inv;
    eshift[o] = b_out[o] * inv + beta[o] - rmean[o] * inv;
  }
}

// ---------------- K1: QKV projection, P = Wcat(384x256) * x(256x2048) ----------------
// grid = B * (N/64); block = 256 (8 waves); wave w owns M rows [48w,48w+48), 4 N-tiles
__global__ __launch_bounds__(256) void k1_qkv(
    const float* __restrict__ x, const _Float16* __restrict__ Wcat,
    const float* __restrict__ bias_cat, _Float16* __restrict__ theta_t,
    _Float16* __restrict__ phi, _Float16* __restrict__ gmat) {
  constexpr int LDSN = 264;  // 256 + 8 pad (keeps 16B alignment, breaks bank conflicts)
  __shared__ _Float16 xs[64 * LDSN];
  const int b   = blockIdx.x >> 5;
  const int n0w = (blockIdx.x & 31) << 6;
  const int tid = threadIdx.x;

  // stage x tile [k=256][n=64] transposed into LDS as [n][k] f16 (float4 reads)
  const float* xb = x + ((size_t)b * 256) * 2048 + n0w;
  for (int it = 0; it < 16; ++it) {
    int idx = tid + (it << 8);           // 0..4095, covers 256k x 16 quads
    int k = idx >> 4, n4 = (idx & 15) << 2;
    v4f q = *(const v4f*)(xb + (size_t)k * 2048 + n4);
    xs[(n4 + 0) * LDSN + k] = (_Float16)q[0];
    xs[(n4 + 1) * LDSN + k] = (_Float16)q[1];
    xs[(n4 + 2) * LDSN + k] = (_Float16)q[2];
    xs[(n4 + 3) * LDSN + k] = (_Float16)q[3];
  }
  __syncthreads();

  const int wave = tid >> 5, lane = tid & 31;
  const int m0 = wave * 48;
  v8f acc[3][4];
  for (int t = 0; t < 3; ++t)
    for (int nt = 0; nt < 4; ++nt) acc[t][nt] = zero8();

  for (int kt = 0; kt < 8; ++kt) {
    const int k0 = kt * 32;
    v16h a[3];
    for (int t = 0; t < 3; ++t) a[t] = load_a(Wcat, m0 + t * 16, k0, 256);
    for (int nt = 0; nt < 4; ++nt) {
      v16h bb = load_b(xs, nt * 16, k0, LDSN);
      for (int t = 0; t < 3; ++t) acc[t][nt] = wmma16(a[t], bb, acc[t][nt]);
    }
  }

  const int hl = lane >> 4, nl = lane & 15;
  for (int t = 0; t < 3; ++t) {
    const int j0 = m0 + t * 16;
    float bs[8];
    for (int i = 0; i < 8; ++i) bs[i] = bias_cat[j0 + 8 * hl + i];
    for (int nt = 0; nt < 4; ++nt) {
      const int n = n0w + nt * 16 + nl;
      if (j0 < 128) {  // theta, stored transposed [n][ci] -> one 16B store
        v8h outv;
        for (int i = 0; i < 8; ++i) outv[i] = (_Float16)(acc[t][nt][i] + bs[i]);
        *(v8h*)(theta_t + ((size_t)b * 2048 + n) * 128 + j0 + 8 * hl) = outv;
      } else {         // phi / g, stored [ci][n]
        _Float16* base2 = (j0 < 256) ? phi : gmat;
        const int joff = (j0 < 256) ? (j0 - 128) : (j0 - 256);
        for (int i = 0; i < 8; ++i) {
          int row = joff + 8 * hl + i;
          base2[((size_t)b * 128 + row) * 2048 + n] = (_Float16)(acc[t][nt][i] + bs[i]);
        }
      }
    }
  }
}

// ---------------- K2: partial S over n-chunk; Spart[b][chunk][j][i] (f32, transposed) ----------
// grid = B * 8 (n split into 8 chunks of 256); wave w owns i-tile w x all 8 j-tiles
__global__ __launch_bounds__(256) void k2_s(
    const _Float16* __restrict__ phi, const _Float16* __restrict__ gmat,
    float* __restrict__ Spart) {
  const int b = blockIdx.x >> 3;
  const int chunk = blockIdx.x & 7;
  const int wave = threadIdx.x >> 5, lane = threadIdx.x & 31;
  const _Float16* pb = phi  + (size_t)b * 128 * 2048;
  const _Float16* gb = gmat + (size_t)b * 128 * 2048;
  const int i0 = wave * 16;
  const int nbase = chunk * 256;
  v8f acc[8];
  for (int jt = 0; jt < 8; ++jt) acc[jt] = zero8();

  for (int n0 = nbase; n0 < nbase + 256; n0 += 32) {
    v16h a = load_a(pb, i0, n0, 2048);
    for (int jt = 0; jt < 8; ++jt) {
      v16h bb = load_b(gb, jt * 16, n0, 2048);
      acc[jt] = wmma16(a, bb, acc[jt]);
    }
  }

  const int hl = lane >> 4, nl = lane & 15;
  float* sp = Spart + ((size_t)(b * 8 + chunk) << 14);  // 128*128 per partial
  for (int jt = 0; jt < 8; ++jt) {
    const int j = jt * 16 + nl;
    *(v8f*)(sp + j * 128 + i0 + 8 * hl) = acc[jt];  // transposed [j][i], 32B store
  }
}

// ---------------- K2b: reduce 8 partials -> S_t[b][c][k] f16, scaled by 1/N ----------------
__global__ __launch_bounds__(256) void k2b_reduce(
    const float* __restrict__ Spart, _Float16* __restrict__ S_t) {
  int idx = blockIdx.x * 256 + threadIdx.x;  // 16*128*128 = 262144
  if (idx >= 16 * 128 * 128) return;
  int b = idx >> 14;
  int ji = idx & 16383;
  float s = 0.0f;
  for (int c = 0; c < 8; ++c) s += Spart[(((size_t)b * 8 + c) << 14) + ji];
  S_t[idx] = (_Float16)(s * (1.0f / 2048.0f));
}

// ---------------- K3: yt[n][c] = sum_k theta_t[n][k] * S_t[c][k] ----------------
// grid = B * (N/128); wave w owns n-tile w x all 8 c-tiles; K = 128
__global__ __launch_bounds__(256) void k3_y(
    const _Float16* __restrict__ theta_t, const _Float16* __restrict__ S_t,
    _Float16* __restrict__ yt) {
  const int b = blockIdx.x >> 4;
  const int nblk = (blockIdx.x & 15) << 7;
  const int wave = threadIdx.x >> 5, lane = threadIdx.x & 31;
  const _Float16* tb = theta_t + ((size_t)b * 2048 + nblk) * 128;
  const _Float16* sb = S_t + (size_t)b * 128 * 128;
  const int n0 = wave * 16;
  v8f acc[8];
  for (int ct = 0; ct < 8; ++ct) acc[ct] = zero8();

  for (int k0 = 0; k0 < 128; k0 += 32) {
    v16h a = load_a(tb, n0, k0, 128);
    for (int ct = 0; ct < 8; ++ct) {
      v16h bb = load_b(sb, ct * 16, k0, 128);
      acc[ct] = wmma16(a, bb, acc[ct]);
    }
  }

  const int hl = lane >> 4, nl = lane & 15;
  for (int ct = 0; ct < 8; ++ct) {
    const int c = ct * 16 + nl;
    for (int i = 0; i < 8; ++i) {
      const int n = nblk + n0 + i + 8 * hl;
      yt[((size_t)b * 2048 + n) * 128 + c] = (_Float16)acc[ct][i];
    }
  }
}

// ---------------- K4: out[o][n] = sum_c Wout[o][c]*yt[n][c], BN epilogue + residual --------
// grid = B * (C/128) * (N/64); wave w owns o-tile (oblk + 16w) x 4 n-tiles; K = 128
__global__ __launch_bounds__(256) void k4_out(
    const _Float16* __restrict__ Wout16, const _Float16* __restrict__ yt,
    const float* __restrict__ escale, const float* __restrict__ eshift,
    const float* __restrict__ x, float* __restrict__ out) {
  const int gx = blockIdx.x;
  const int b = gx >> 6;
  const int r = gx & 63;
  const int oblk = (r >> 5) << 7;
  const int n0w = (r & 31) << 6;
  const int wave = threadIdx.x >> 5, lane = threadIdx.x & 31;
  const int o0 = oblk + wave * 16;
  const _Float16* yb = yt + ((size_t)b * 2048 + n0w) * 128;
  v8f acc[4];
  for (int nt = 0; nt < 4; ++nt) acc[nt] = zero8();

  for (int k0 = 0; k0 < 128; k0 += 32) {
    v16h a = load_a(Wout16, o0, k0, 128);
    for (int nt = 0; nt < 4; ++nt) {
      v16h bb = load_b(yb, nt * 16, k0, 128);
      acc[nt] = wmma16(a, bb, acc[nt]);
    }
  }

  const int hl = lane >> 4, nl = lane & 15;
  float sc[8], sh[8];
  for (int i = 0; i < 8; ++i) {
    int o = o0 + i + 8 * hl;
    sc[i] = escale[o];
    sh[i] = eshift[o];
  }
  for (int nt = 0; nt < 4; ++nt) {
    const int n = n0w + nt * 16 + nl;
    for (int i = 0; i < 8; ++i) {
      const int o = o0 + i + 8 * hl;
      const size_t idx = ((size_t)b * 256 + o) * 2048 + n;
      out[idx] = acc[nt][i] * sc[i] + sh[i] + x[idx];
    }
  }
}

extern "C" void kernel_launch(void* const* d_in, const int* in_sizes, int n_in,
                              void* d_out, int out_size, void* d_ws, size_t ws_size,
                              hipStream_t stream) {
  const float* x       = (const float*)d_in[0];
  const float* w_theta = (const float*)d_in[1];
  const float* b_theta = (const float*)d_in[2];
  const float* w_phi   = (const float*)d_in[3];
  const float* b_phi   = (const float*)d_in[4];
  const float* w_g     = (const float*)d_in[5];
  const float* b_g     = (const float*)d_in[6];
  const float* w_out   = (const float*)d_in[7];
  const float* b_out   = (const float*)d_in[8];
  const float* gamma   = (const float*)d_in[9];
  const float* beta    = (const float*)d_in[10];
  const float* rmean   = (const float*)d_in[11];
  const float* rvar    = (const float*)d_in[12];
  float* out = (float*)d_out;

  char* ws = (char*)d_ws;
  size_t off = 0;
  auto alloc = [&](size_t bytes) {
    void* p = ws + off;
    off = (off + bytes + 255) & ~(size_t)255;
    return p;
  };
  _Float16* Wcat    = (_Float16*)alloc((size_t)384 * 256 * 2);
  _Float16* Wout16  = (_Float16*)alloc((size_t)256 * 128 * 2);
  float*    bias_c  = (float*)alloc(384 * 4);
  float*    escale  = (float*)alloc(256 * 4);
  float*    eshift  = (float*)alloc(256 * 4);
  _Float16* theta_t = (_Float16*)alloc((size_t)16 * 2048 * 128 * 2);
  _Float16* phi     = (_Float16*)alloc((size_t)16 * 2048 * 128 * 2);
  _Float16* gmat    = (_Float16*)alloc((size_t)16 * 2048 * 128 * 2);
  float*    Spart   = (float*)alloc((size_t)16 * 8 * 128 * 128 * 4);
  _Float16* S_t     = (_Float16*)alloc((size_t)16 * 128 * 128 * 2);
  _Float16* yt      = (_Float16*)alloc((size_t)16 * 2048 * 128 * 2);

  k0_prep<<<515, 256, 0, stream>>>(w_theta, w_phi, w_g, b_theta, b_phi, b_g,
                                   w_out, b_out, gamma, beta, rmean, rvar,
                                   Wcat, Wout16, bias_c, escale, eshift);
  k1_qkv<<<512, 256, 0, stream>>>(x, Wcat, bias_c, theta_t, phi, gmat);
  k2_s<<<128, 256, 0, stream>>>(phi, gmat, Spart);
  k2b_reduce<<<1024, 256, 0, stream>>>(Spart, S_t);
  k3_y<<<256, 256, 0, stream>>>(theta_t, S_t, yt);
  k4_out<<<1024, 256, 0, stream>>>(Wout16, yt, escale, eshift, x, out);
}
